// AttnResBlock_9131100471745
// MI455X (gfx1250) — matrix operand order repaired and checked by reading, not compiled, not run on hardware
//
#include <hip/hip_runtime.h>
#include <hip/hip_bf16.h>

// ---------------------------------------------------------------------------
// AttnResBlock for MI455X (gfx1250).
// All contractions via v_wmma_f32_16x16x32_bf16, bf16 operands staged
// K-contiguous. Each wave computes a 32x32 tile (2x2 register block, 4
// accumulators) with a software-pipelined K loop: 8 global_load_b128 feed 4
// WMMAs per step, next step's fragments loaded before the current WMMAs issue.
// B=16, C=512, A=64, L=1024, fp32 I/O.
// ---------------------------------------------------------------------------

#define BB 16
#define CC 512
#define AA 64
#define LL 1024
#define LP (LL + 2)          // zero-padded length for k=3 conv
#define EPSBN 1e-5f

typedef __attribute__((ext_vector_type(16))) __bf16 v16bf;
typedef __attribute__((ext_vector_type(8)))  __bf16 v8bf;
typedef __attribute__((ext_vector_type(8)))  float  v8f;

static __device__ __forceinline__ __bf16 f2bf(float f) {
  union { float f; unsigned u; } in; in.f = f;
  unsigned r = in.u + 0x7FFFu + ((in.u >> 16) & 1u);   // round-to-nearest-even
  union { unsigned short s; __bf16 b; } out;
  out.s = (unsigned short)(r >> 16);
  return out.b;
}

// Fragment loads (wave32 layouts):
//   A: lane m=lane&15; elems j<8 -> K=(lane>>4)*8+j, j>=8 -> +16
//   B: lane n=lane&15; elems j<8 -> K=(lane>>4)*8+j, j>=8 -> +16
static __device__ __forceinline__ v16bf load_afrag(const __bf16* row, int k0, int half) {
  v8bf lo = *(const v8bf*)(row + k0 + half * 8);
  v8bf hi = *(const v8bf*)(row + k0 + half * 8 + 16);
  v16bf r;
#pragma unroll
  for (int j = 0; j < 8; ++j) { r[j] = lo[j]; r[j + 8] = hi[j]; }
  return r;
}
static __device__ __forceinline__ v16bf load_bfrag(const __bf16* row, int k0, int half) {
  v8bf lo = *(const v8bf*)(row + k0 + half * 8);
  v8bf hi = *(const v8bf*)(row + k0 + half * 8 + 16);
  v16bf r;
#pragma unroll
  for (int j = 0; j < 8; ++j) { r[j] = lo[j]; r[j + 8] = hi[j]; }
  return r;
}
#define WMMA_BF16(acc, a, b) \
  __builtin_amdgcn_wmma_f32_16x16x32_bf16(false, (a), false, (b), (short)0, (acc), false, false)

// ---------------------------------------------------------------------------
// Generic batched bf16 GEMM: D[b][m,n] = sum_k A[b][m,k] * B[b][k,n]
//   A: bf16 m-major, K contiguous, element (m,k) at m*lda + k
//   B: bf16 n-major, K contiguous, element (k,n) at n*ldb + k
// Epilogue: +bias[m], +resid (f32, m*ldr+n); writes f32 normal (Dn) and/or
// bf16 transposed (Dt, element (m,n) at n*ldt+m, vectorized 8xbf16 stores).
// One 32x32 output tile per wave (2x2 WMMA register block).
// ---------------------------------------------------------------------------
__global__ __launch_bounds__(256)
void gemm_bf16_wmma(const __bf16* __restrict__ A, const __bf16* __restrict__ Bp,
                    const float* __restrict__ bias, const float* __restrict__ resid,
                    float* __restrict__ Dn, __bf16* __restrict__ Dt,
                    int N, int K, int lda, int ldb, long sAb, long sBb,
                    int ldd, long sDb, int ldr, long sRb, int ldt, long sTb) {
  const int b     = blockIdx.y;
  const int wave  = threadIdx.x >> 5;
  const int lane  = threadIdx.x & 31;
  const int half  = lane >> 4;
  const int idx16 = lane & 15;

  const int tiles_n = N >> 5;                 // 32-wide tiles
  const int tile    = blockIdx.x * 8 + wave;
  const int tile_m  = tile / tiles_n;
  const int tile_n  = tile % tiles_n;

  const __bf16* Ar0 = A  + (long)b * sAb + ((long)tile_m * 32 + idx16) * lda;
  const __bf16* Ar1 = Ar0 + (long)16 * lda;
  const __bf16* Br0 = Bp + (long)b * sBb + ((long)tile_n * 32 + idx16) * ldb;
  const __bf16* Br1 = Br0 + (long)16 * ldb;

  v8f acc00 = {}, acc01 = {}, acc10 = {}, acc11 = {};
  v16bf a0 = load_afrag(Ar0, 0, half), a1 = load_afrag(Ar1, 0, half);
  v16bf b0 = load_bfrag(Br0, 0, half), b1 = load_bfrag(Br1, 0, half);

  int k0 = 0;
  for (; k0 + 32 < K; k0 += 32) {             // software-pipelined
    const int kn = k0 + 32;
    v16bf na0 = load_afrag(Ar0, kn, half), na1 = load_afrag(Ar1, kn, half);
    v16bf nb0 = load_bfrag(Br0, kn, half), nb1 = load_bfrag(Br1, kn, half);
    __builtin_prefetch(Br0 + kn + 32, 0, 1);  // global_prefetch_b8
    acc00 = WMMA_BF16(acc00, a0, b0);
    acc01 = WMMA_BF16(acc01, a0, b1);
    acc10 = WMMA_BF16(acc10, a1, b0);
    acc11 = WMMA_BF16(acc11, a1, b1);
    a0 = na0; a1 = na1; b0 = nb0; b1 = nb1;
  }
  acc00 = WMMA_BF16(acc00, a0, b0);
  acc01 = WMMA_BF16(acc01, a0, b1);
  acc10 = WMMA_BF16(acc10, a1, b0);
  acc11 = WMMA_BF16(acc11, a1, b1);

  const v8f accs[4] = {acc00, acc01, acc10, acc11};
  const int mloc = half * 8;
#pragma unroll
  for (int si = 0; si < 2; ++si) {
#pragma unroll
    for (int sj = 0; sj < 2; ++sj) {
      const v8f acc = accs[si * 2 + sj];
      const int n = tile_n * 32 + sj * 16 + idx16;
      const int mbase = tile_m * 32 + si * 16 + mloc;
      float v[8];
#pragma unroll
      for (int r = 0; r < 8; ++r) {
        v[r] = acc[r];
        if (bias)  v[r] += bias[mbase + r];
        if (resid) v[r] += resid[(long)b * sRb + (long)(mbase + r) * ldr + n];
      }
      if (Dn) {
#pragma unroll
        for (int r = 0; r < 8; ++r)
          Dn[(long)b * sDb + (long)(mbase + r) * ldd + n] = v[r];
      }
      if (Dt) {
        v8bf ot;
#pragma unroll
        for (int r = 0; r < 8; ++r) ot[r] = f2bf(v[r]);
        *(v8bf*)(Dt + (long)b * sTb + (long)n * ldt + mbase) = ot;
      }
    }
  }
}

// ---------------------------------------------------------------------------
// Conv1d k=3 pad=1: D[b][o,l] = bias[o] + sum_t sum_c Wt[t][o,c]*HpT[b][l+t,c]
//   Wt:  3 tap-split C x C bf16 matrices, K contiguous
//   HpT: (B, L+2, C) bf16, zero rows at l=0 and l=L+1, K (=c) contiguous
// 32x32 tile per wave; flattened pipelined loop over 3 taps x C/32 K-steps.
// ---------------------------------------------------------------------------
__global__ __launch_bounds__(256)
void conv3_bf16_wmma(const __bf16* __restrict__ Wt, const __bf16* __restrict__ HpT,
                     const float* __restrict__ bias, const float* __restrict__ resid,
                     float* __restrict__ D) {
  const int b     = blockIdx.y;
  const int wave  = threadIdx.x >> 5;
  const int lane  = threadIdx.x & 31;
  const int half  = lane >> 4;
  const int idx16 = lane & 15;

  const int tile   = blockIdx.x * 8 + wave;
  const int tile_m = tile >> 5;        // C/32 = 16 row tiles
  const int tile_n = tile & 31;        // L/32 = 32 col tiles

  const __bf16* Arow = Wt  + ((long)tile_m * 32 + idx16) * CC;     // + t*CC*CC
  const __bf16* Brow = HpT + (long)b * LP * CC
                           + ((long)tile_n * 32 + idx16) * CC;     // + (t + 16*sj)*CC

  const int NSTEP = 3 * (CC / 32);     // 48
  v8f acc00 = {}, acc01 = {}, acc10 = {}, acc11 = {};

  // step s: t = s/16, k0 = (s%16)*32
  v16bf a0 = load_afrag(Arow, 0, half);
  v16bf a1 = load_afrag(Arow + 16L * CC, 0, half);
  v16bf b0 = load_bfrag(Brow, 0, half);
  v16bf b1 = load_bfrag(Brow + 16L * CC, 0, half);

  for (int s = 0; s + 1 < NSTEP; ++s) {
    const int sn = s + 1;
    const int tn = sn >> 4, kn = (sn & 15) << 5;
    const __bf16* An = Arow + (long)tn * CC * CC;
    const __bf16* Bn = Brow + (long)tn * CC;
    v16bf na0 = load_afrag(An, kn, half);
    v16bf na1 = load_afrag(An + 16L * CC, kn, half);
    v16bf nb0 = load_bfrag(Bn, kn, half);
    v16bf nb1 = load_bfrag(Bn + 16L * CC, kn, half);
    acc00 = WMMA_BF16(acc00, a0, b0);
    acc01 = WMMA_BF16(acc01, a0, b1);
    acc10 = WMMA_BF16(acc10, a1, b0);
    acc11 = WMMA_BF16(acc11, a1, b1);
    a0 = na0; a1 = na1; b0 = nb0; b1 = nb1;
  }
  acc00 = WMMA_BF16(acc00, a0, b0);
  acc01 = WMMA_BF16(acc01, a0, b1);
  acc10 = WMMA_BF16(acc10, a1, b0);
  acc11 = WMMA_BF16(acc11, a1, b1);

  const v8f accs[4] = {acc00, acc01, acc10, acc11};
  const int mloc = half * 8;
#pragma unroll
  for (int si = 0; si < 2; ++si) {
#pragma unroll
    for (int sj = 0; sj < 2; ++sj) {
      const v8f acc = accs[si * 2 + sj];
      const int n = tile_n * 32 + sj * 16 + idx16;
      const int mbase = tile_m * 32 + si * 16 + mloc;
#pragma unroll
      for (int r = 0; r < 8; ++r) {
        int m = mbase + r;
        float v = acc[r] + bias[m];
        long off = ((long)b * CC + m) * LL + n;
        if (resid) v += resid[off];
        D[off] = v;
      }
    }
  }
}

// ---------------------------------------------------------------------------
// Row softmax over last axis with scale 1/(L/2); one block per (b,l) row.
// ---------------------------------------------------------------------------
__global__ __launch_bounds__(256)
void softmax_kernel(float* __restrict__ S) {
  float* row = S + (long)blockIdx.x * LL;
  __shared__ float red[256];
  const int t = threadIdx.x;
  const float isc = 1.0f / 512.0f;          // scores / (seqlen/2)

  float v0 = row[t] * isc, v1 = row[t + 256] * isc,
        v2 = row[t + 512] * isc, v3 = row[t + 768] * isc;
  float mx = fmaxf(fmaxf(v0, v1), fmaxf(v2, v3));
  red[t] = mx; __syncthreads();
  for (int s = 128; s > 0; s >>= 1) { if (t < s) red[t] = fmaxf(red[t], red[t + s]); __syncthreads(); }
  mx = red[0]; __syncthreads();

  float e0 = __expf(v0 - mx), e1 = __expf(v1 - mx),
        e2 = __expf(v2 - mx), e3 = __expf(v3 - mx);
  red[t] = e0 + e1 + e2 + e3; __syncthreads();
  for (int s = 128; s > 0; s >>= 1) { if (t < s) red[t] += red[t + s]; __syncthreads(); }
  float inv = 1.0f / red[0];

  row[t] = e0 * inv; row[t + 256] = e1 * inv; row[t + 512] = e2 * inv; row[t + 768] = e3 * inv;
}

// ---------------------------------------------------------------------------
// LDS-tiled transpose + f32->bf16 convert: in (R x Cn) f32 -> out (Cn x R) bf16.
// ---------------------------------------------------------------------------
__global__ __launch_bounds__(256)
void transpose_cvt_kernel(const float* __restrict__ in, __bf16* __restrict__ out,
                          int R, int Cn) {
  __shared__ float tile[32][33];
  const int c0 = blockIdx.x * 32, r0 = blockIdx.y * 32;
  const float* ib = in + (long)blockIdx.z * R * Cn;
  __bf16* ob = out + (long)blockIdx.z * R * Cn;
  const int tx = threadIdx.x & 31, ty = threadIdx.x >> 5;
#pragma unroll
  for (int i = 0; i < 32; i += 8)
    tile[ty + i][tx] = ib[(long)(r0 + ty + i) * Cn + c0 + tx];
  __syncthreads();
#pragma unroll
  for (int i = 0; i < 32; i += 8)
    ob[(long)(c0 + ty + i) * R + r0 + tx] = f2bf(tile[tx][ty + i]);
}

// Flat f32 -> bf16 convert.
__global__ __launch_bounds__(256)
void cvt_bf16_kernel(const float* __restrict__ in, __bf16* __restrict__ out) {
  long i = (long)blockIdx.x * 256 + threadIdx.x;
  out[i] = f2bf(in[i]);
}

// Tap-split conv weights: out[t*C*C + o*C + c] = W[o*3C + c*3 + t], bf16.
__global__ __launch_bounds__(256)
void tapsplit_kernel(const float* __restrict__ W, __bf16* __restrict__ out) {
  long i = (long)blockIdx.x * 256 + threadIdx.x;   // over 3*C*C
  int t = (int)(i / (CC * CC));
  int rem = (int)(i % (CC * CC));
  int o = rem / CC, c = rem % CC;
  out[i] = f2bf(W[(long)o * (3 * CC) + (long)c * 3 + t]);
}

// ---------------------------------------------------------------------------
// BatchNorm statistics: per channel c, mean & rsqrt(var+eps) over (B, L).
// ---------------------------------------------------------------------------
__global__ __launch_bounds__(256)
void bnstats_kernel(const float* __restrict__ Z, float* __restrict__ mean,
                    float* __restrict__ istd) {
  const int c = blockIdx.x;
  const int t = threadIdx.x;
  float s = 0.f, ss = 0.f;
  for (int b = 0; b < BB; ++b) {
    const float* p = Z + ((long)b * CC + c) * LL;
    for (int l = t; l < LL; l += 256) { float v = p[l]; s += v; ss += v * v; }
  }
  __shared__ float r0[256], r1[256];
  r0[t] = s; r1[t] = ss; __syncthreads();
  for (int k = 128; k > 0; k >>= 1) {
    if (t < k) { r0[t] += r0[t + k]; r1[t] += r1[t + k]; }
    __syncthreads();
  }
  if (t == 0) {
    const float inv_n = 1.0f / (float)(BB * LL);
    float m = r0[0] * inv_n;
    float var = r1[0] * inv_n - m * m;
    mean[c] = m;
    istd[c] = rsqrtf(var + EPSBN);
  }
}

// ---------------------------------------------------------------------------
// h = relu(bn(z)) -> transposed, zero-padded bf16 (B, L+2, C) for the convs.
// ---------------------------------------------------------------------------
__global__ __launch_bounds__(256)
void bnrelu_padT_kernel(const float* __restrict__ Z, const float* __restrict__ mean,
                        const float* __restrict__ istd, const float* __restrict__ gamma,
                        const float* __restrict__ beta, __bf16* __restrict__ HpT) {
  long idx = (long)blockIdx.x * 256 + threadIdx.x;   // over B*(L+2)*C, c fastest
  int c = (int)(idx % CC);
  int l = (int)((idx / CC) % LP);
  long b = idx / ((long)CC * LP);
  float out = 0.f;
  if (l >= 1 && l <= LL) {
    float v = Z[((long)b * CC + c) * LL + (l - 1)];
    v = (v - mean[c]) * istd[c] * gamma[c] + beta[c];
    out = fmaxf(v, 0.f);
  }
  HpT[idx] = f2bf(out);
}

// ---------------------------------------------------------------------------
// Workspace layout (bytes), total ~169 MB:
//  [0,64M)    S (B,L,L) f32  -> reused: y f32 at 0 (32M), h2 f32 at 32M (32M)
//  [64,96M)   attnT (B,L,L) bf16
//  [96,112M)  xT_bf (B,L,C) bf16        [112,128M) x_bf (B,C,L) bf16
//  [128,130M) keysT (B,L,A) bf16        [130,132M) queriesT bf16
//  [132,148M) xatT (B,L,C) bf16
//  [148,165M) HpT (B,L+2,C) bf16 (h1pT then h3pT)
//  [165M...)  bf16 weights: Wk,Wq,Wp,W1t,W2t ; [169M) BN stats f32
// ---------------------------------------------------------------------------
extern "C" void kernel_launch(void* const* d_in, const int* in_sizes, int n_in,
                              void* d_out, int out_size, void* d_ws, size_t ws_size,
                              hipStream_t stream) {
  (void)in_sizes; (void)n_in; (void)out_size; (void)ws_size;
  const float* x  = (const float*)d_in[0];
  const float* Wk = (const float*)d_in[1];
  const float* bk = (const float*)d_in[2];
  const float* Wq = (const float*)d_in[3];
  const float* bq = (const float*)d_in[4];
  const float* Wp = (const float*)d_in[5];
  const float* bp = (const float*)d_in[6];
  const float* W1 = (const float*)d_in[7];
  const float* b1 = (const float*)d_in[8];
  const float* W2 = (const float*)d_in[9];
  const float* b2 = (const float*)d_in[10];
  const float* g1 = (const float*)d_in[11];
  const float* be1 = (const float*)d_in[12];
  const float* g2 = (const float*)d_in[13];
  const float* be2 = (const float*)d_in[14];

  const long MBL = 1024L * 1024L;
  char* base = (char*)d_ws;
  float*  S     = (float*)(base);
  float*  y     = (float*)(base);              // reuse after S is dead
  float*  h2    = (float*)(base + 32 * MBL);
  __bf16* attnT = (__bf16*)(base + 64 * MBL);
  __bf16* xT    = (__bf16*)(base + 96 * MBL);
  __bf16* xbf   = (__bf16*)(base + 112 * MBL);
  __bf16* keysT = (__bf16*)(base + 128 * MBL);
  __bf16* querT = (__bf16*)(base + 130 * MBL);
  __bf16* xatT  = (__bf16*)(base + 132 * MBL);
  __bf16* HpT   = (__bf16*)(base + 148 * MBL);
  __bf16* Wkb   = (__bf16*)(base + 165 * MBL);           // 64K elems
  __bf16* Wqb   = Wkb + 64 * 1024;
  __bf16* Wpb   = Wqb + 64 * 1024;                       // 256K elems
  __bf16* W1t   = Wpb + 256 * 1024;                      // 768K elems
  __bf16* W2t   = W1t + 768 * 1024;
  float*  stats = (float*)(base + 169 * MBL);
  float *mean1 = stats, *istd1 = stats + 512, *mean2 = stats + 1024, *istd2 = stats + 1536;
  float* out = (float*)d_out;

  dim3 blk(256);

  // --- stage bf16 operands -------------------------------------------------
  cvt_bf16_kernel<<<dim3((BB * CC * LL) / 256), blk, 0, stream>>>(x, xbf);
  transpose_cvt_kernel<<<dim3(LL / 32, CC / 32, BB), blk, 0, stream>>>(x, xT, CC, LL);
  cvt_bf16_kernel<<<dim3((AA * CC) / 256), blk, 0, stream>>>(Wk, Wkb);
  cvt_bf16_kernel<<<dim3((AA * CC) / 256), blk, 0, stream>>>(Wq, Wqb);
  cvt_bf16_kernel<<<dim3((CC * CC) / 256), blk, 0, stream>>>(Wp, Wpb);
  tapsplit_kernel<<<dim3((3 * CC * CC) / 256), blk, 0, stream>>>(W1, W1t);
  tapsplit_kernel<<<dim3((3 * CC * CC) / 256), blk, 0, stream>>>(W2, W2t);

  // --- attention -----------------------------------------------------------
  // keysT[l,a] = (Wk @ x + bk)^T      (M=64, N=1024, K=512): 64 tiles -> 8 blocks
  gemm_bf16_wmma<<<dim3(8, BB), blk, 0, stream>>>(
      Wkb, xT, bk, nullptr, nullptr, keysT, LL, CC,
      CC, CC, 0L, (long)LL * CC, 0, 0L, 0, 0L, AA, (long)LL * AA);
  gemm_bf16_wmma<<<dim3(8, BB), blk, 0, stream>>>(
      Wqb, xT, bq, nullptr, nullptr, querT, LL, CC,
      CC, CC, 0L, (long)LL * CC, 0, 0L, 0, 0L, AA, (long)LL * AA);
  // S[l,m] = sum_a keysT[l,a] * querT[m,a]   (M=N=1024, K=64): 1024 tiles
  gemm_bf16_wmma<<<dim3(128, BB), blk, 0, stream>>>(
      keysT, querT, nullptr, nullptr, S, nullptr, LL, AA,
      AA, AA, (long)LL * AA, (long)LL * AA, LL, (long)LL * LL, 0, 0L, 0, 0L);
  // attn = softmax(S / 512) rows, then transpose-convert to bf16
  softmax_kernel<<<dim3(BB * LL), blk, 0, stream>>>(S);
  transpose_cvt_kernel<<<dim3(LL / 32, LL / 32, BB), blk, 0, stream>>>(S, attnT, LL, LL);
  // xatT[l,c] = (x @ attn)^T          (M=512, N=1024, K=1024): 512 tiles
  gemm_bf16_wmma<<<dim3(64, BB), blk, 0, stream>>>(
      xbf, attnT, nullptr, nullptr, nullptr, xatT, LL, LL,
      LL, LL, (long)CC * LL, (long)LL * LL, 0, 0L, 0, 0L, CC, (long)LL * CC);
  // y = Wp @ xat + bp + x   (attention residual; S region now dead)
  gemm_bf16_wmma<<<dim3(64, BB), blk, 0, stream>>>(
      Wpb, xatT, bp, x, y, nullptr, LL, CC,
      CC, CC, 0L, (long)LL * CC, LL, (long)CC * LL, LL, (long)CC * LL, 0, 0L);

  // --- residual block ------------------------------------------------------
  const int padBlocks = (BB * CC * LP) / 256;   // 32832
  bnstats_kernel<<<dim3(CC), blk, 0, stream>>>(y, mean1, istd1);
  bnrelu_padT_kernel<<<dim3(padBlocks), blk, 0, stream>>>(y, mean1, istd1, g1, be1, HpT);
  conv3_bf16_wmma<<<dim3(64, BB), blk, 0, stream>>>(W1t, HpT, b1, nullptr, h2);
  bnstats_kernel<<<dim3(CC), blk, 0, stream>>>(h2, mean2, istd2);
  bnrelu_padT_kernel<<<dim3(padBlocks), blk, 0, stream>>>(h2, mean2, istd2, g2, be2, HpT);
  conv3_bf16_wmma<<<dim3(64, BB), blk, 0, stream>>>(W2t, HpT, b2, y, out);
}
